// Net_57767310131657
// MI455X (gfx1250) — compile-verified, matrix operand order checked by reading, not run on hardware
//
#include <hip/hip_runtime.h>
#include <hip/hip_bf16.h>

typedef __attribute__((ext_vector_type(16))) _Float16 v16h;
typedef __attribute__((ext_vector_type(8)))  _Float16 v8h;
typedef __attribute__((ext_vector_type(8)))  float    v8f;

// ---------------- constants for the GZ2 decision-tree head ----------------
__constant__ int c_qb[12]  = {0,3,5,7,9,13,15,18,25,28,31,37};
__constant__ int c_seg[37] = {0,0,0, 1,1, 2,2, 3,3, 4,4,4,4, 5,5, 6,6,6,
                              7,7,7,7,7,7,7, 8,8,8, 9,9,9, 10,10,10,10,10,10};
__constant__ int c_depcnt[11]    = {0,1,2,2,2,0,1,1,2,3,3};
__constant__ int c_depidx[11][3] = {{0,0,0},{1,0,0},{1,4,0},{1,4,0},{1,4,0},
                                    {0,0,0},{0,0,0},{13,0,0},{1,3,0},{1,4,7},{1,4,7}};

// ---------------- sizes ----------------
#define NB     512
#define H1_N   (NB*16*30*30)          // 7,372,800 f16
#define W2P_N  (8*13*32*16)           // 53,248 f16   [nt][kc][lane][16]
#define R2_N   (NB*128*676)           // pre-pool conv2, f16
#define H2_N   (NB*21632)             // flattened pooled conv2, f16
#define W1P_N  (16*676*32*16)         // 5,537,792 f16 [nt][kc][lane][16]
#define H3_N   (NB*256)               // fc1 output, f16

// per-lane K offset inside a 32-wide K chunk for the f16 WMMA fragment layout
__device__ __forceinline__ int frag_koff(int lane, int j) {
    return ((j < 8) ? j : j + 8) + ((lane >= 16) ? 8 : 0);
}

// ================= conv1 (direct) + ReLU + 2x2 maxpool =================
// x[512,3,64,64] f32, w[16,3,5,5], b[16] -> h1[512,16,30,30] f16
__global__ void k_conv1(const float* __restrict__ x, const float* __restrict__ w,
                        const float* __restrict__ bias, _Float16* __restrict__ h1) {
    int idx = blockIdx.x * blockDim.x + threadIdx.x;
    if (idx >= NB * 16 * 30 * 30) return;
    int px = idx % 30; int t = idx / 30;
    int py = t % 30;   t /= 30;
    int c  = t % 16;   int b = t / 16;
    const float* xb = x + (size_t)b * 3 * 64 * 64;
    const float* wc = w + c * 75;
    float bi = bias[c];
    float best = 0.f;                          // ReLU output is >= 0
    #pragma unroll
    for (int dy = 0; dy < 2; ++dy)
    #pragma unroll
    for (int dx = 0; dx < 2; ++dx) {
        int oy = 2 * py + dy, ox = 2 * px + dx;
        float acc = bi;
        for (int ci = 0; ci < 3; ++ci)
            #pragma unroll
            for (int ky = 0; ky < 5; ++ky) {
                const float* row = xb + (ci * 64 + oy + ky) * 64 + ox;
                const float* wr  = wc + ci * 25 + ky * 5;
                #pragma unroll
                for (int kx = 0; kx < 5; ++kx) acc += row[kx] * wr[kx];
            }
        best = fmaxf(best, acc);
    }
    h1[idx] = (_Float16)best;
}

// ================= weight repack: conv2 -> WMMA B fragments =================
// conv2_w[128,16,5,5]  ->  w2p[(nt*13+kc)*32+lane][16], K padded 400->416
__global__ void k_prep_w2(const float* __restrict__ w, _Float16* __restrict__ w2p) {
    int idx = blockIdx.x * blockDim.x + threadIdx.x;
    if (idx >= W2P_N) return;
    int j    = idx & 15;
    int lane = (idx >> 4) & 31;
    int kc   = (idx >> 9) % 13;
    int nt   = idx / (13 * 512);
    int n    = nt * 16 + (lane & 15);
    int kk   = kc * 32 + frag_koff(lane, j);
    float v  = (kk < 400) ? w[n * 400 + kk] : 0.f;   // flat [n][c*25+ky*5+kx]
    w2p[idx] = (_Float16)v;
}

// ================= weight repack: fc1 -> WMMA B fragments =================
// fc1_w[256,21632] -> w1p[(nt*676+kc)*32+lane][16]   (K = 676*32 exactly)
__global__ void k_prep_w1(const float* __restrict__ w, _Float16* __restrict__ w1p) {
    int idx = blockIdx.x * blockDim.x + threadIdx.x;
    if (idx >= W1P_N) return;
    int j    = idx & 15;
    int lane = (idx >> 4) & 31;
    int kc   = (idx >> 9) % 676;
    int nt   = idx / (676 * 512);
    int n    = nt * 16 + (lane & 15);
    int kk   = kc * 32 + frag_koff(lane, j);
    w1p[idx] = (_Float16)w[(size_t)n * 21632 + kk];
}

// ================= conv2 implicit GEMM via WMMA =================
// per image: [676 x 416(im2col)] x [416 x 128] ; grid (512, 43), 8 waves = 8 N-tiles
__global__ __launch_bounds__(256) void k_conv2(const _Float16* __restrict__ h1,
                                               const _Float16* __restrict__ w2p,
                                               const float* __restrict__ bias,
                                               _Float16* __restrict__ r2) {
    __shared__ __align__(16) _Float16 As[16 * 416];   // 13 KB im2col A-tile
    int b   = blockIdx.x;
    int mt  = blockIdx.y;
    int tid = threadIdx.x;

    const _Float16* h1b = h1 + (size_t)b * 16 * 900;
    for (int i = tid; i < 16 * 416; i += 256) {
        int m  = i / 416, kk = i % 416;
        int p  = mt * 16 + m;
        _Float16 v = (_Float16)0.f;
        if (p < 676 && kk < 400) {
            int oy = p / 26, ox = p % 26;
            int c  = kk / 25, rem = kk % 25;
            int ky = rem / 5, kx = rem % 5;
            v = h1b[(c * 30 + oy + ky) * 30 + ox + kx];
        }
        As[i] = v;
    }
    __syncthreads();

    int wave = tid >> 5;                // N-tile id (0..7)
    int lane = tid & 31;
    int mrow = lane & 15;
    int khalf = (lane >> 4) * 8;

    const _Float16* Arow  = As + mrow * 416 + khalf;
    const _Float16* Bbase = w2p + wave * (13 * 512) + lane * 16;

    v8f acc = {};
    #pragma unroll
    for (int kc = 0; kc < 13; ++kc) {
        v8h a0 = *(const v8h*)(Arow + kc * 32);        // K +0..7 of this half
        v8h a1 = *(const v8h*)(Arow + kc * 32 + 16);   // K +16..23 of this half
        v16h av;
        #pragma unroll
        for (int i = 0; i < 8; ++i) { av[i] = a0[i]; av[i + 8] = a1[i]; }
        v16h bv = *(const v16h*)(Bbase + kc * 512);
        acc = __builtin_amdgcn_wmma_f32_16x16x32_f16(false, av, false, bv,
                                                     (short)0, acc, false, false);
    }

    int n  = wave * 16 + (lane & 15);
    float bi = bias[n];
    int mb = mt * 16 + ((lane >> 4) ? 8 : 0);
    _Float16* rr = r2 + ((size_t)b * 128 + n) * 676;
    #pragma unroll
    for (int i = 0; i < 8; ++i) {
        int p = mb + i;
        if (p < 676) rr[p] = (_Float16)fmaxf(acc[i] + bi, 0.f);
    }
}

// ================= 2x2 maxpool of ReLU'd conv2 + flatten =================
__global__ void k_pool2(const _Float16* __restrict__ r2, _Float16* __restrict__ h2) {
    int idx = blockIdx.x * blockDim.x + threadIdx.x;
    if (idx >= NB * 128 * 169) return;
    int x13 = idx % 13; int t = idx / 13;
    int y13 = t % 13;   t /= 13;
    int c   = t % 128;  int b = t / 128;
    const _Float16* base = r2 + ((size_t)b * 128 + c) * 676;
    int p = (2 * y13) * 26 + 2 * x13;
    float v = fmaxf(fmaxf((float)base[p],      (float)base[p + 1]),
                    fmaxf((float)base[p + 26], (float)base[p + 27]));
    h2[idx] = (_Float16)v;   // idx == b*21632 + c*169 + y13*13 + x13
}

// ================= fc1 GEMM via WMMA, split-K over 8 waves =================
// [512 x 21632] x [21632 x 256]; grid (32 mtiles, 16 ntiles)
__global__ __launch_bounds__(256) void k_fc1(const _Float16* __restrict__ h2,
                                             const _Float16* __restrict__ w1p,
                                             const float* __restrict__ bias,
                                             _Float16* __restrict__ h3) {
    __shared__ float red[8][16][16];
    int mt = blockIdx.x, nt = blockIdx.y;
    int wave = threadIdx.x >> 5, lane = threadIdx.x & 31;
    int mrow = lane & 15;
    int khalf = (lane >> 4) * 8;

    const _Float16* Abase = h2 + ((size_t)(mt * 16 + mrow)) * 21632 + khalf;
    const _Float16* Bbase = w1p + (size_t)nt * (676 * 512) + lane * 16;

    v8f acc = {};
    for (int kc = wave; kc < 676; kc += 8) {
        __builtin_prefetch(Bbase + (kc + 8) * 512, 0, 0);  // global_prefetch_b8
        v8h a0 = *(const v8h*)(Abase + kc * 32);
        v8h a1 = *(const v8h*)(Abase + kc * 32 + 16);
        v16h av;
        #pragma unroll
        for (int i = 0; i < 8; ++i) { av[i] = a0[i]; av[i + 8] = a1[i]; }
        v16h bv = *(const v16h*)(Bbase + kc * 512);
        acc = __builtin_amdgcn_wmma_f32_16x16x32_f16(false, av, false, bv,
                                                     (short)0, acc, false, false);
    }

    int mhi = (lane >> 4) * 8;
    #pragma unroll
    for (int i = 0; i < 8; ++i) red[wave][mhi + i][mrow] = acc[i];
    __syncthreads();

    int m_l = threadIdx.x >> 4;   // 0..15
    int n_l = threadIdx.x & 15;   // 0..15
    float s = 0.f;
    #pragma unroll
    for (int w = 0; w < 8; ++w) s += red[w][m_l][n_l];
    int n = nt * 16 + n_l;
    s = fmaxf(s + bias[n], 0.f);
    h3[(size_t)(mt * 16 + m_l) * 256 + n] = (_Float16)s;
}

// ====== fc2 + sigmoid + segment normalize + dependency products ======
__global__ __launch_bounds__(64) void k_head(const _Float16* __restrict__ h3,
                                             const float* __restrict__ w2,
                                             const float* __restrict__ b2,
                                             float* __restrict__ out) {
    __shared__ float ys[37];
    __shared__ float qsum[11];
    __shared__ float nrm[37];
    int b = blockIdx.x, t = threadIdx.x;
    if (t < 37) {
        const _Float16* hr = h3 + (size_t)b * 256;
        const float*    wr = w2 + t * 256;
        float acc = b2[t];
        #pragma unroll 8
        for (int k = 0; k < 256; ++k) acc += (float)hr[k] * wr[k];
        ys[t] = 1.f / (1.f + __expf(-acc));
    }
    __syncthreads();
    if (t < 11) {
        float s = 0.f;
        for (int a = c_qb[t]; a < c_qb[t + 1]; ++a) s += ys[a];
        qsum[t] = s + 1e-12f;
    }
    __syncthreads();
    if (t < 37) nrm[t] = ys[t] / qsum[c_seg[t]];
    __syncthreads();
    if (t < 37) {
        int q = c_seg[t];
        float f = 1.f;
        for (int j = 0; j < c_depcnt[q]; ++j) f *= nrm[c_depidx[q][j]];
        out[b * 37 + t] = nrm[t] * f;
    }
}

// ================= launch =================
extern "C" void kernel_launch(void* const* d_in, const int* in_sizes, int n_in,
                              void* d_out, int out_size, void* d_ws, size_t ws_size,
                              hipStream_t stream) {
    const float* x       = (const float*)d_in[0];
    const float* conv1_w = (const float*)d_in[1];
    const float* conv1_b = (const float*)d_in[2];
    const float* conv2_w = (const float*)d_in[3];
    const float* conv2_b = (const float*)d_in[4];
    const float* fc1_w   = (const float*)d_in[5];
    const float* fc1_b   = (const float*)d_in[6];
    const float* fc2_w   = (const float*)d_in[7];
    const float* fc2_b   = (const float*)d_in[8];
    float* out = (float*)d_out;

    char* ws = (char*)d_ws;
    size_t off = 0;
    _Float16* h1  = (_Float16*)(ws + off); off += (size_t)H1_N  * 2;
    _Float16* w2p = (_Float16*)(ws + off); off += (size_t)W2P_N * 2;
    _Float16* r2  = (_Float16*)(ws + off); off += (size_t)R2_N  * 2;
    _Float16* h2  = (_Float16*)(ws + off); off += (size_t)H2_N  * 2;
    _Float16* w1p = (_Float16*)(ws + off); off += (size_t)W1P_N * 2;
    _Float16* h3  = (_Float16*)(ws + off); off += (size_t)H3_N  * 2;

    // weight repack (independent of activations)
    k_prep_w2<<<(W2P_N + 255) / 256, 256, 0, stream>>>(conv2_w, w2p);
    k_prep_w1<<<(W1P_N + 255) / 256, 256, 0, stream>>>(fc1_w, w1p);

    // conv1 + relu + pool
    k_conv1<<<(NB * 16 * 900 + 255) / 256, 256, 0, stream>>>(x, conv1_w, conv1_b, h1);

    // conv2 implicit GEMM (WMMA) + relu
    k_conv2<<<dim3(NB, 43), 256, 0, stream>>>(h1, w2p, conv2_b, r2);

    // pool + flatten
    k_pool2<<<(NB * 128 * 169 + 255) / 256, 256, 0, stream>>>(r2, h2);

    // fc1 GEMM (WMMA, split-K) + relu
    k_fc1<<<dim3(32, 16), 256, 0, stream>>>(h2, w1p, fc1_b, h3);

    // fc2 + sigmoid + GZ2 tree head
    k_head<<<NB, 64, 0, stream>>>(h3, fc2_w, fc2_b, out);
}
